// ContextPythiaAttention_12154757447839
// MI455X (gfx1250) — matrix operand order repaired and checked by reading, not compiled, not run on hardware
//
#include <hip/hip_runtime.h>
#include <hip/hip_bf16.h>

typedef _Float16 h16;
typedef __attribute__((ext_vector_type(16))) _Float16 v16h;
typedef __attribute__((ext_vector_type(8)))  _Float16 v8h;
typedef __attribute__((ext_vector_type(8)))  float    v8f;

#define BATCH 2
#define SEQ 2048
#define CTX 512
#define HIDDEN 1024
#define NHEAD 16
#define HDIM 64
#define NTOK (BATCH * SEQ)          // 4096
#define QKVN (3 * HIDDEN)           // 3072

// ---------- fragment helpers (ISA 7.12.2 16-bit layouts, wave32) ----------

__device__ inline v16h combine8(v8h lo, v8h hi) {
  return __builtin_shufflevector(lo, hi, 0,1,2,3,4,5,6,7,8,9,10,11,12,13,14,15);
}

// A tile 16x32 f16 from row-major [.., lda]
__device__ inline v16h load_a_frag(const h16* __restrict__ base, int lda, int lane) {
  int m  = lane & 15;
  int kb = (lane >> 4) * 8;
  const h16* p = base + (size_t)m * lda;
  v8h lo = *(const v8h*)(p + kb);
  v8h hi = *(const v8h*)(p + 16 + kb);
  return combine8(lo, hi);
}

// B tile 32x16 f16 where B[k,n] = BT[n,k]; base at BT(n0,k0), row stride ldb
__device__ inline v16h load_b_frag(const h16* __restrict__ base, int ldb, int lane) {
  int n  = lane & 15;
  int kb = (lane >> 4) * 8;
  const h16* p = base + (size_t)n * ldb + kb;
  v8h lo = *(const v8h*)(p);
  v8h hi = *(const v8h*)(p + 16);
  return combine8(lo, hi);
}

__device__ inline v8f wmma_f16(v16h a, v16h b, v8f c) {
  return __builtin_amdgcn_wmma_f32_16x16x32_f16(false, a, false, b, (short)0, c,
                                                false, false);
}

// xor-butterfly max within 16-lane rows via v_permlane16_b32 (VALU, no DS stalls)
__device__ inline float maxshfl16(float v, unsigned s0, unsigned s1) {
  union { float f; unsigned u; } a, b;
  a.f = v;
  b.u = __builtin_amdgcn_permlane16(a.u, a.u, s0, s1, false, false);
  return fmaxf(v, b.f);
}

// async global->LDS 16B-per-lane copy (CDNA5, tracked by ASYNCcnt)
#define ASYNC_B128(ldsoff, gaddr)                                         \
  asm volatile("global_load_async_to_lds_b128 %0, %1, off"                \
               :: "v"(ldsoff), "v"(gaddr) : "memory")
#define WAIT_ASYNC() asm volatile("s_wait_asynccnt 0x0" ::: "memory")

// ---------- conversion kernels ----------

__global__ void cvt_f32_to_f16(const float* __restrict__ src, h16* __restrict__ dst, int n) {
  int i = blockIdx.x * blockDim.x + threadIdx.x;
  if (i < n) dst[i] = (h16)src[i];
}

__global__ void cvt_transpose_f16(const float* __restrict__ src, h16* __restrict__ dst,
                                  int rows, int cols) {
  int i = blockIdx.x * blockDim.x + threadIdx.x;
  if (i >= rows * cols) return;
  int k = i / cols, n = i - k * cols;
  dst[(size_t)n * rows + k] = (h16)src[(size_t)k * cols + n];
}

// ---------- WMMA GEMM: one wave -> 32x64 output, 8 wmma per 32-deep k-step ----------

template <typename OutT>
__global__ void gemm_wmma(const h16* __restrict__ A, const h16* __restrict__ BT,
                          const float* __restrict__ bias, OutT* __restrict__ C,
                          int M, int N, int K) {
  int lane = threadIdx.x & 31;
  int wid  = blockIdx.x * (blockDim.x >> 5) + (threadIdx.x >> 5);
  int nstrips = N >> 6;
  int mb = wid / nstrips;
  int ns = wid - mb * nstrips;
  int m0 = mb * 32, n0 = ns * 64;
  if (m0 >= M) return;

  v8f acc[2][4] = {};
  for (int k0 = 0; k0 < K; k0 += 32) {
    v16h a0 = load_a_frag(A + (size_t)m0 * K + k0, K, lane);
    v16h a1 = load_a_frag(A + (size_t)(m0 + 16) * K + k0, K, lane);
#pragma unroll
    for (int t = 0; t < 4; ++t) {
      v16h b = load_b_frag(BT + (size_t)(n0 + t * 16) * K + k0, K, lane);
      acc[0][t] = wmma_f16(a0, b, acc[0][t]);
      acc[1][t] = wmma_f16(a1, b, acc[1][t]);
    }
  }
  int hi = lane >> 4, nl = lane & 15;
#pragma unroll
  for (int half = 0; half < 2; ++half)
#pragma unroll
    for (int t = 0; t < 4; ++t) {
      int col = n0 + t * 16 + nl;
      float bv = bias[col];
#pragma unroll
      for (int r = 0; r < 8; ++r) {
        int row = m0 + half * 16 + r + 8 * hi;
        C[(size_t)row * N + col] = (OutT)(acc[half][t][r] + bv);
      }
    }
}

// ---------- rotary embedding (first 16 dims of q,k per head), in-place f16 ----------

__global__ void rope_kernel(h16* __restrict__ qkv) {
  int idx = blockIdx.x * blockDim.x + threadIdx.x;   // token*NHEAD + head
  if (idx >= NTOK * NHEAD) return;
  int token = idx >> 4, head = idx & 15;
  int s = token & (SEQ - 1);
  h16* base = qkv + (size_t)token * QKVN + head * (3 * HDIM);
#pragma unroll
  for (int j = 0; j < 8; ++j) {
    float inv = __expf(-(float)j * 1.1512925465f);   // 10000^(-j/8)
    float ang = (float)s * inv;
    float c = cosf(ang), sn = sinf(ang);
    float qa = (float)base[j], qb = (float)base[j + 8];
    base[j]     = (h16)(qa * c - qb * sn);
    base[j + 8] = (h16)(qb * c + qa * sn);
    float ka = (float)base[HDIM + j], kb = (float)base[HDIM + j + 8];
    base[HDIM + j]     = (h16)(ka * c - kb * sn);
    base[HDIM + j + 8] = (h16)(kb * c + ka * sn);
  }
}

// ---------- V transpose: qkv v-slice -> vT[(b*NHEAD+h)*HDIM + d][s] ----------

__global__ void vtrans_kernel(const h16* __restrict__ qkv, h16* __restrict__ vT) {
  int idx = blockIdx.x * blockDim.x + threadIdx.x;   // over NTOK * HIDDEN
  if (idx >= NTOK * HIDDEN) return;
  int token = idx >> 10;
  int hd = idx & 1023;
  int head = hd >> 6, d = hd & 63;
  int b = token >> 11, s = token & (SEQ - 1);
  h16 v = qkv[(size_t)token * QKVN + head * (3 * HDIM) + 2 * HDIM + d];
  vT[((size_t)(b * NHEAD + head) * HDIM + d) * SEQ + s] = v;
}

// ---------- flash attention: one wave per (b,h,16-row q-tile) ----------
// K/V tiles staged in per-wave LDS slices via async global->LDS copies.

__global__ void attn_kernel(const h16* __restrict__ qkv, const h16* __restrict__ vT,
                            h16* __restrict__ aout) {
  __shared__ __align__(16) h16 kbuf[4][32 * 64];  // [key][d]
  __shared__ __align__(16) h16 vbuf[4][64 * 32];  // [d][key]
  __shared__ __align__(16) h16 pbuf[4][16 * 32];  // row-major P

  int lane = threadIdx.x & 31;
  int wv = threadIdx.x >> 5;
  int wid = blockIdx.x * 4 + wv;
  int itile = wid & 127;            // SEQ/16 q-tiles per (b,h)
  int bh = wid >> 7;                // 0..31
  int b = bh >> 4, h = bh & 15;
  int m0 = itile * 16;
  int lanem = lane & 15, hi = lane >> 4;
  const float scale = 0.125f;       // 1/sqrt(64)

  const h16* qbase = qkv + ((size_t)(b * SEQ + m0)) * QKVN + h * (3 * HDIM);
  const h16* kbase = qkv + ((size_t)(b * SEQ)) * QKVN + h * (3 * HDIM) + HDIM;
  const h16* vtb   = vT + (size_t)bh * HDIM * SEQ;
  const h16* k_lds = &kbuf[wv][0];
  const h16* v_lds = &vbuf[wv][0];
  h16* p_lds = &pbuf[wv][0];
  unsigned ldsK = (unsigned)(uintptr_t)k_lds;
  unsigned ldsV = (unsigned)(uintptr_t)v_lds;

  v16h aQ0 = load_a_frag(qbase, QKVN, lane);        // d 0..31
  v16h aQ1 = load_a_frag(qbase + 32, QKVN, lane);   // d 32..63
  v16h bOnes;
#pragma unroll
  for (int j = 0; j < 16; ++j) bOnes[j] = (h16)1.0f;

  float mrow[8];
  v8f O[4] = {};
  v8f Lacc = {};
#pragma unroll
  for (int r = 0; r < 8; ++r) mrow[r] = -1e30f;

  int nblocks = (m0 + 16 + 31) >> 5;
  for (int kb = 0; kb < nblocks; ++kb) {
    int nstart = kb * 32;

    // ---- async stage: K tile 32x64 (lane = key row), V tile 64x32 (lane = d row)
    {
      unsigned long long gK =
          (unsigned long long)(uintptr_t)(kbase + (size_t)(nstart + lane) * QKVN);
      unsigned lK = ldsK + lane * 128;
#pragma unroll
      for (int c = 0; c < 8; ++c) ASYNC_B128(lK + c * 16, gK + c * 16);
      unsigned long long gV0 =
          (unsigned long long)(uintptr_t)(vtb + (size_t)lane * SEQ + nstart);
      unsigned long long gV1 = gV0 + (unsigned long long)32 * SEQ * 2;
      unsigned lV0 = ldsV + lane * 64;
      unsigned lV1 = lV0 + 2048;
#pragma unroll
      for (int c = 0; c < 4; ++c) {
        ASYNC_B128(lV0 + c * 16, gV0 + c * 16);
        ASYNC_B128(lV1 + c * 16, gV1 + c * 16);
      }
      WAIT_ASYNC();
    }

    // ---- S = Q K^T (fragments out of LDS)
    v8f s[2] = {};
#pragma unroll
    for (int t = 0; t < 2; ++t) {
      v16h b0 = load_b_frag(k_lds + t * 16 * 64, 64, lane);       // d 0..31
      v16h b1 = load_b_frag(k_lds + t * 16 * 64 + 32, 64, lane);  // d 32..63
      s[t] = wmma_f16(aQ0, b0, s[t]);
      s[t] = wmma_f16(aQ1, b1, s[t]);
    }

    // ---- online softmax (max only; row-sum comes from an extra WMMA)
    float p0[8], p1[8];
#pragma unroll
    for (int r = 0; r < 8; ++r) {
      int row = m0 + r + 8 * hi;
      float v0 = s[0][r] * scale;
      float v1 = s[1][r] * scale;
      if (nstart + lanem > row)      v0 = -1e30f;
      if (nstart + 16 + lanem > row) v1 = -1e30f;
      float mx = fmaxf(v0, v1);
      mx = maxshfl16(mx, 0x67452301u, 0xEFCDAB89u);   // xor 1
      mx = maxshfl16(mx, 0x54761032u, 0xDCFE98BAu);   // xor 2
      mx = maxshfl16(mx, 0x32107654u, 0xBA98FEDCu);   // xor 4
      mx = maxshfl16(mx, 0xFEDCBA98u, 0x76543210u);   // xor 8
      float mnew = fmaxf(mrow[r], mx);
      float alpha = __expf(mrow[r] - mnew);
      mrow[r] = mnew;
      p0[r] = __expf(v0 - mnew);
      p1[r] = __expf(v1 - mnew);
#pragma unroll
      for (int t = 0; t < 4; ++t) O[t][r] *= alpha;
      Lacc[r] *= alpha;
    }

    // ---- P (C-layout) -> row-major LDS -> A fragment
#pragma unroll
    for (int r = 0; r < 8; ++r) {
      p_lds[(r + 8 * hi) * 32 + lanem]      = (h16)p0[r];
      p_lds[(r + 8 * hi) * 32 + 16 + lanem] = (h16)p1[r];
    }
    asm volatile("" ::: "memory");   // DS ops of a wave are in-order; just pin ordering
    v16h aP = load_a_frag(p_lds, 32, lane);
    asm volatile("" ::: "memory");

    // ---- denominator: L += P * ones   (row sums land in C-layout like O)
    Lacc = wmma_f16(aP, bOnes, Lacc);

    // ---- O += P V
#pragma unroll
    for (int t = 0; t < 4; ++t) {
      v16h bV = load_b_frag(v_lds + t * 16 * 32, 32, lane);
      O[t] = wmma_f16(aP, bV, O[t]);
    }
  }

  // ---- epilogue: normalize, write [token][h*64+d] as f16
#pragma unroll
  for (int t = 0; t < 4; ++t)
#pragma unroll
    for (int r = 0; r < 8; ++r) {
      int row = m0 + r + 8 * hi;
      aout[((size_t)(b * SEQ + row)) * HIDDEN + h * HDIM + t * 16 + lanem] =
          (h16)(O[t][r] / Lacc[r]);
    }
}

// ---------- host launcher ----------

extern "C" void kernel_launch(void* const* d_in, const int* in_sizes, int n_in,
                              void* d_out, int out_size, void* d_ws, size_t ws_size,
                              hipStream_t stream) {
  (void)in_sizes; (void)n_in; (void)out_size; (void)ws_size;
  const float* ctx  = (const float*)d_in[0];
  const float* Wqkv = (const float*)d_in[1];
  const float* bqkv = (const float*)d_in[2];
  const float* Wd   = (const float*)d_in[3];
  const float* bd   = (const float*)d_in[4];
  float* out = (float*)d_out;

  char* w = (char*)d_ws;
  const size_t SZ_CTX = (size_t)NTOK * CTX * 2;
  const size_t SZ_WQ  = (size_t)QKVN * CTX * 2;
  const size_t SZ_WD  = (size_t)HIDDEN * HIDDEN * 2;
  const size_t SZ_QKV = (size_t)NTOK * QKVN * 2;
  const size_t SZ_VT  = (size_t)BATCH * NHEAD * HDIM * SEQ * 2;
  h16* ctx_h = (h16*)(w);
  h16* wqkvT = (h16*)(w + SZ_CTX);
  h16* wdT   = (h16*)(w + SZ_CTX + SZ_WQ);
  h16* qkv_h = (h16*)(w + SZ_CTX + SZ_WQ + SZ_WD);
  h16* vT    = (h16*)(w + SZ_CTX + SZ_WQ + SZ_WD + SZ_QKV);
  h16* aout  = (h16*)(w + SZ_CTX + SZ_WQ + SZ_WD + SZ_QKV + SZ_VT);

  {
    int n = NTOK * CTX;
    cvt_f32_to_f16<<<(n + 255) / 256, 256, 0, stream>>>(ctx, ctx_h, n);
  }
  {
    int n = CTX * QKVN;
    cvt_transpose_f16<<<(n + 255) / 256, 256, 0, stream>>>(Wqkv, wqkvT, CTX, QKVN);
  }
  {
    int n = HIDDEN * HIDDEN;
    cvt_transpose_f16<<<(n + 255) / 256, 256, 0, stream>>>(Wd, wdT, HIDDEN, HIDDEN);
  }

  // QKV GEMM: [4096,512] x [512,3072] + bqkv -> f16
  {
    int waves = (NTOK / 32) * (QKVN / 64);   // 128*48 = 6144
    gemm_wmma<h16><<<waves / 4, 128, 0, stream>>>(ctx_h, wqkvT, bqkv, qkv_h,
                                                  NTOK, QKVN, CTX);
  }

  {
    int n = NTOK * NHEAD;
    rope_kernel<<<(n + 255) / 256, 256, 0, stream>>>(qkv_h);
  }
  {
    int n = NTOK * HIDDEN;
    vtrans_kernel<<<(n + 255) / 256, 256, 0, stream>>>(qkv_h, vT);
  }

  // flash attention: 4096 waves, 4 per block
  attn_kernel<<<1024, 128, 0, stream>>>(qkv_h, vT, aout);

  // output projection: [4096,1024] x [1024,1024] + bd -> f32
  {
    int waves = (NTOK / 32) * (HIDDEN / 64); // 128*16 = 2048
    gemm_wmma<float><<<waves / 4, 128, 0, stream>>>(aout, wdT, bd, out,
                                                    NTOK, HIDDEN, HIDDEN);
  }
}
